// MultiHeadAttention_25804163514741
// MI455X (gfx1250) — matrix operand order repaired and checked by reading, not compiled, not run on hardware
//
#include <hip/hip_runtime.h>

// MHA forward for B=2, S=2048, D=1024, H=16, HD=64 on gfx1250 (MI455X).
// bf16 WMMA for all matmuls; flash softmax; bf16 intermediates in d_ws (48 MB,
// L2-resident). GEMMs stage the shared B panel in LDS via CDNA5 async
// global->LDS copies (ASYNCcnt), double-buffered against the WMMA stream.

typedef __attribute__((ext_vector_type(16))) __bf16 v16bf;
typedef __attribute__((ext_vector_type(8)))  float  v8f;
typedef __attribute__((ext_vector_type(4)))  unsigned int u32x4;
typedef __attribute__((ext_vector_type(4)))  float  f32x4;
typedef __attribute__((ext_vector_type(4)))  unsigned short u16x4;
typedef unsigned short u16;

#define BATCH 2
#define SEQ   2048
#define DIM   1024
#define HEADS 16
#define HD    64

union Frag { v16bf v; u32x4 q[2]; };

__device__ __forceinline__ u16 f2bf(float x) {
  unsigned u = __float_as_uint(x);
  unsigned r = (u + 0x7FFFu + ((u >> 16) & 1u)) >> 16;
  return (u16)r;
}

// A-fragment: 16x32 bf16 (MxK), row-major src, leading dim lda.
// Lane l: row = l&15; K-chunks [8*hi..8*hi+7] and [16+8*hi..23+8*hi].
__device__ __forceinline__ v16bf load_afrag(const u16* base, int lda, int lane) {
  int r = lane & 15, hi = lane >> 4;
  const u16* p = base + r * lda + 8 * hi;
  Frag f;
  f.q[0] = *(const u32x4*)(p);
  f.q[1] = *(const u32x4*)(p + 16);
  return f.v;
}

// B-fragment: 32x16 bf16 (KxN) supplied as B^T row-major (N-major, ldb).
// Lane l: col n = l&15; K-chunks [8*hi..8*hi+7] and [16+8*hi..23+8*hi].
__device__ __forceinline__ v16bf load_bfrag(const u16* bt, int ldb, int lane) {
  int n = lane & 15, hi = lane >> 4;
  const u16* p = bt + n * ldb + 8 * hi;
  Frag f;
  f.q[0] = *(const u32x4*)(p);
  f.q[1] = *(const u32x4*)(p + 16);
  return f.v;
}

__device__ __forceinline__ v8f wmma_bf16(v16bf a, v16bf b, v8f c) {
  return __builtin_amdgcn_wmma_f32_16x16x32_bf16(
      false, a, false, b, (short)0, c, false, false);
}

// Async global->LDS b128 copy (CDNA5, ASYNCcnt-tracked). Inline asm so it is
// portable across ROCm-7.2 / amdgpu-toolchain builtin arity differences.
__device__ __forceinline__ void async_b128(const u16* gaddr, const u16* lds_ptr) {
  unsigned lds_off = (unsigned)(size_t)lds_ptr;  // AS3 offset lives in low 32 bits
  asm volatile("global_load_async_to_lds_b128 %0, %1, off"
               :: "v"(lds_off), "v"(gaddr) : "memory");
}
__device__ __forceinline__ void wait_async0() {
  asm volatile("s_wait_asynccnt 0x0" ::: "memory");
}

// Stage one 64x32 bf16 B panel (N-major) into LDS: 256 threads x 16B.
__device__ __forceinline__ void stage_panel(const u16* wT, int n0, int k0,
                                            u16* panel, int tid) {
  int row = tid >> 2;        // 0..63 (n within panel)
  int ch  = tid & 3;         // 16-byte chunk within 64B row
  async_b128(wT + (n0 + row) * DIM + k0 + ch * 8, panel + row * 32 + ch * 8);
}

// ---------------- Stage 1: fp32 -> bf16 convert (vectorized) ----------------
__global__ void __launch_bounds__(256) conv_bf16_kernel(const float* __restrict__ src,
                                                        u16* __restrict__ dst) {
  int i = (blockIdx.x * 256 + threadIdx.x) * 4;
  f32x4 v = *(const f32x4*)(src + i);
  u16x4 o;
  o.x = f2bf(v.x); o.y = f2bf(v.y); o.z = f2bf(v.z); o.w = f2bf(v.w);
  *(u16x4*)(dst + i) = o;
}

// ---------------- Stage 2: transpose fp32 [K][N] -> bf16 [N][K] ----------------
__global__ void __launch_bounds__(256) transpose_bf16_kernel(const float* __restrict__ src,
                                                             u16* __restrict__ dst,
                                                             int K, int N) {
  int i = blockIdx.x * 256 + threadIdx.x;   // over N*K, coalesced on dst
  int n = i / K;
  int kk = i - n * K;
  dst[i] = f2bf(src[kk * N + n]);
}

// ---------------- Stage 3: QKV GEMM ----------------
// 256 threads = 8 waves; block tile 256x64; wave tile 32x64 (8 WMMAs/k-step).
// B panel (64x32) staged in LDS by async copy, double-buffered.
__global__ void __launch_bounds__(256) qkv_gemm_kernel(const u16* __restrict__ xb,
                                                       const u16* __restrict__ wT,
                                                       const float* __restrict__ bias,
                                                       u16* __restrict__ qo,
                                                       u16* __restrict__ ko,
                                                       u16* __restrict__ vTo) {
  __shared__ u16 panels[2][64 * 32];
  int tid = threadIdx.x;
  int lane = tid & 31, wave = tid >> 5;
  int m0 = blockIdx.x * 256 + wave * 32;
  int n0 = blockIdx.y * 64;

  stage_panel(wT, n0, 0, &panels[0][0], tid);

  v8f acc0[4] = {}, acc1[4] = {};
  for (int kk = 0; kk < DIM / 32; ++kk) {
    int k0 = kk * 32;
    wait_async0();
    __syncthreads();                      // panel[kk&1] ready for every wave
    if (kk + 1 < DIM / 32)
      stage_panel(wT, n0, k0 + 32, &panels[(kk + 1) & 1][0], tid);  // overlaps WMMAs
    const u16* pan = &panels[kk & 1][0];
    v16bf a0 = load_afrag(xb + (m0 +  0) * DIM + k0, DIM, lane);
    v16bf a1 = load_afrag(xb + (m0 + 16) * DIM + k0, DIM, lane);
    v16bf b[4];
#pragma unroll
    for (int t = 0; t < 4; ++t) b[t] = load_bfrag(pan + t * 16 * 32, 32, lane);
#pragma unroll
    for (int t = 0; t < 4; ++t) {
      acc0[t] = wmma_bf16(a0, b[t], acc0[t]);
      acc1[t] = wmma_bf16(a1, b[t], acc1[t]);
    }
  }

  int hi = lane >> 4, nl = lane & 15;
#pragma unroll
  for (int t = 0; t < 4; ++t) {
    int n = n0 + t * 16 + nl;              // column in 3D; layout h*(3*HD)+e
    int h = n / (3 * HD);
    int e = n - h * (3 * HD);
    float bv = bias[n];
#pragma unroll
    for (int g = 0; g < 2; ++g) {
      const v8f* acc = g ? acc1 : acc0;
#pragma unroll
      for (int j = 0; j < 8; ++j) {
        int m = m0 + g * 16 + j + 8 * hi;
        int b_ = m >> 11;                  // / SEQ
        int s  = m & (SEQ - 1);
        u16 w = f2bf(acc[t][j] + bv);
        int head = b_ * HEADS + h;
        if (e < HD)
          qo[(head * SEQ + s) * HD + e] = w;
        else if (e < 2 * HD)
          ko[(head * SEQ + s) * HD + (e - HD)] = w;
        else
          vTo[(head * HD + (e - 2 * HD)) * SEQ + s] = w;  // transposed store
      }
    }
  }
}

// ---------------- Stage 4: flash attention ----------------
// One wave per 16-query tile per (b,h). Per 32-key chunk: K-frags issued first,
// V-frags second (in-order completion keeps V in flight through softmax).
__global__ void __launch_bounds__(32) attn_kernel(const u16* __restrict__ q,
                                                  const u16* __restrict__ k,
                                                  const u16* __restrict__ vT,
                                                  u16* __restrict__ attn_out) {
  __shared__ u16 lds_p[16 * 32];           // P tile bounce: C-layout -> A-layout

  int lane = threadIdx.x & 31;
  int hi = lane >> 4, nl = lane & 15;
  int qt = blockIdx.x;                     // 0..S/16-1
  int bh = blockIdx.y;                     // 0..B*H-1

  const u16* qh = q  + (size_t)bh * SEQ * HD;
  const u16* kh = k  + (size_t)bh * SEQ * HD;
  const u16* vh = vT + (size_t)bh * HD * SEQ;

  v16bf aq0 = load_afrag(qh + qt * 16 * HD + 0,  HD, lane);
  v16bf aq1 = load_afrag(qh + qt * 16 * HD + 32, HD, lane);

  float M[8], L[8];
  v8f o[4] = {};
#pragma unroll
  for (int j = 0; j < 8; ++j) { M[j] = -1e30f; L[j] = 0.0f; }

  for (int kc = 0; kc < SEQ; kc += 32) {
    // K fragments for the 16x32 score tile (issued first)
    v16bf bk0 = load_bfrag(kh + (kc + 0)  * HD + 0,  HD, lane);
    v16bf bk1 = load_bfrag(kh + (kc + 0)  * HD + 32, HD, lane);
    v16bf bk2 = load_bfrag(kh + (kc + 16) * HD + 0,  HD, lane);
    v16bf bk3 = load_bfrag(kh + (kc + 16) * HD + 32, HD, lane);
    // V fragments (issued second; stay outstanding through softmax)
    v16bf bv[4];
#pragma unroll
    for (int dt = 0; dt < 4; ++dt)
      bv[dt] = load_bfrag(vh + dt * 16 * SEQ + kc, SEQ, lane);

    v8f s0 = {}, s1 = {};
    s0 = wmma_bf16(aq0, bk0, s0);
    s0 = wmma_bf16(aq1, bk1, s0);
    s1 = wmma_bf16(aq0, bk2, s1);
    s1 = wmma_bf16(aq1, bk3, s1);

    // online softmax: C-frag rows spread across 16 lanes -> shfl reductions
#pragma unroll
    for (int j = 0; j < 8; ++j) {
      float x0 = s0[j] * 0.125f;           // 1/sqrt(HD)
      float x1 = s1[j] * 0.125f;
      float mj = fmaxf(x0, x1);
#pragma unroll
      for (int msk = 1; msk < 16; msk <<= 1)
        mj = fmaxf(mj, __shfl_xor(mj, msk, 32));
      float Mn = fmaxf(M[j], mj);
      float corr = __expf(M[j] - Mn);
      float p0 = __expf(x0 - Mn);
      float p1 = __expf(x1 - Mn);
      float rs = p0 + p1;
#pragma unroll
      for (int msk = 1; msk < 16; msk <<= 1)
        rs += __shfl_xor(rs, msk, 32);
      L[j] = L[j] * corr + rs;
      M[j] = Mn;
      o[0][j] *= corr; o[1][j] *= corr; o[2][j] *= corr; o[3][j] *= corr;
      int row = j + 8 * hi;
      lds_p[row * 32 + nl] = f2bf(p0);
      lds_p[row * 32 + 16 + nl] = f2bf(p1);
    }

    // P (16x32) back as A-fragment; accumulate O += P @ V
    v16bf ap = load_afrag((const u16*)lds_p, 32, lane);
#pragma unroll
    for (int dt = 0; dt < 4; ++dt)
      o[dt] = wmma_bf16(ap, bv[dt], o[dt]);
  }

  int b_ = bh >> 4, h = bh & 15;
#pragma unroll
  for (int j = 0; j < 8; ++j) {
    float inv = 1.0f / L[j];
    int s = qt * 16 + j + 8 * hi;
#pragma unroll
    for (int dt = 0; dt < 4; ++dt)
      attn_out[((size_t)b_ * SEQ + s) * DIM + h * HD + dt * 16 + nl] =
          f2bf(o[dt][j] * inv);
  }
}

// ---------------- Stage 5: output projection ----------------
__global__ void __launch_bounds__(256) out_gemm_kernel(const u16* __restrict__ attn,
                                                       const u16* __restrict__ wT,
                                                       const float* __restrict__ bias,
                                                       float* __restrict__ out) {
  __shared__ u16 panels[2][64 * 32];
  int tid = threadIdx.x;
  int lane = tid & 31, wave = tid >> 5;
  int m0 = blockIdx.x * 256 + wave * 32;
  int n0 = blockIdx.y * 64;

  stage_panel(wT, n0, 0, &panels[0][0], tid);

  v8f acc0[4] = {}, acc1[4] = {};
  for (int kk = 0; kk < DIM / 32; ++kk) {
    int k0 = kk * 32;
    wait_async0();
    __syncthreads();
    if (kk + 1 < DIM / 32)
      stage_panel(wT, n0, k0 + 32, &panels[(kk + 1) & 1][0], tid);
    const u16* pan = &panels[kk & 1][0];
    v16bf a0 = load_afrag(attn + (m0 +  0) * DIM + k0, DIM, lane);
    v16bf a1 = load_afrag(attn + (m0 + 16) * DIM + k0, DIM, lane);
    v16bf b[4];
#pragma unroll
    for (int t = 0; t < 4; ++t) b[t] = load_bfrag(pan + t * 16 * 32, 32, lane);
#pragma unroll
    for (int t = 0; t < 4; ++t) {
      acc0[t] = wmma_bf16(a0, b[t], acc0[t]);
      acc1[t] = wmma_bf16(a1, b[t], acc1[t]);
    }
  }

  int hi = lane >> 4, nl = lane & 15;
#pragma unroll
  for (int t = 0; t < 4; ++t) {
    int n = n0 + t * 16 + nl;
    float bv = bias[n];
#pragma unroll
    for (int j = 0; j < 8; ++j) {
      out[(size_t)(m0 + j + 8 * hi) * DIM + n]      = acc0[t][j] + bv;
      out[(size_t)(m0 + 16 + j + 8 * hi) * DIM + n] = acc1[t][j] + bv;
    }
  }
}

extern "C" void kernel_launch(void* const* d_in, const int* in_sizes, int n_in,
                              void* d_out, int out_size, void* d_ws, size_t ws_size,
                              hipStream_t stream) {
  const float* x     = (const float*)d_in[0];
  const float* w_qkv = (const float*)d_in[1];
  const float* b_qkv = (const float*)d_in[2];
  const float* w_out = (const float*)d_in[3];
  const float* b_out = (const float*)d_in[4];
  float* out = (float*)d_out;

  // Workspace layout (bytes), total 48 MB:
  char* ws = (char*)d_ws;
  u16* xb    = (u16*)(ws);                          //  8 MB  x as bf16
  u16* wqkvT = (u16*)(ws + (size_t)8  * 1048576);   //  6 MB  w_qkv^T bf16 [3D][D]
  u16* woutT = (u16*)(ws + (size_t)14 * 1048576);   //  2 MB  w_out^T bf16 [D][D]
  u16* qw    = (u16*)(ws + (size_t)16 * 1048576);   //  8 MB  q [B,H,S,HD]
  u16* kw    = (u16*)(ws + (size_t)24 * 1048576);   //  8 MB  k [B,H,S,HD]
  u16* vTw   = (u16*)(ws + (size_t)32 * 1048576);   //  8 MB  v^T [B,H,HD,S]
  u16* attn  = (u16*)(ws + (size_t)40 * 1048576);   //  8 MB  attn values [B,S,D]

  conv_bf16_kernel<<<(BATCH * SEQ * DIM) / 1024, 256, 0, stream>>>(x, xb);
  transpose_bf16_kernel<<<(3 * DIM * DIM) / 256, 256, 0, stream>>>(w_qkv, wqkvT, DIM, 3 * DIM);
  transpose_bf16_kernel<<<(DIM * DIM) / 256, 256, 0, stream>>>(w_out, woutT, DIM, DIM);
  qkv_gemm_kernel<<<dim3((BATCH * SEQ) / 256, (3 * DIM) / 64), 256, 0, stream>>>(
      xb, wqkvT, b_qkv, qw, kw, vTw);
  attn_kernel<<<dim3(SEQ / 16, BATCH * HEADS), 32, 0, stream>>>(qw, kw, vTw, attn);
  out_gemm_kernel<<<dim3((BATCH * SEQ) / 256, DIM / 64), 256, 0, stream>>>(
      attn, woutT, b_out, out);
}
